// GAT_21861383536795
// MI455X (gfx1250) — matrix operand order repaired and checked by reading, not compiled, not run on hardware
//
#include <hip/hip_runtime.h>
#include <hip/hip_bf16.h>

#define GB 8
#define GN 1024
#define GD 128
#define GE 128
#define GH 4
#define NEG_BIG (-1.0e10f)
#define ALPHA_LRELU 0.2f

typedef __attribute__((ext_vector_type(16))) __bf16 v16bf;
typedef __attribute__((ext_vector_type(2)))  __bf16 v2bf;
typedef __attribute__((ext_vector_type(8)))  float  v8f;

union ABuf { v16bf v; unsigned d[8]; unsigned short u[16]; };
union BBuf { v16bf v; uint4 q[2]; };

__device__ __forceinline__ unsigned short f2bf(float f) {
  unsigned u = __builtin_bit_cast(unsigned, f);
  u = (u + 0x7FFFu + ((u >> 16) & 1u)) >> 16;   // round-to-nearest-even
  return (unsigned short)u;
}

// pack two f32 -> one dword holding two bf16, using the hardware converter
__device__ __forceinline__ unsigned pkbf(float lo, float hi) {
#if __has_builtin(__builtin_amdgcn_cvt_pk_bf16_f32)
  v2bf p = __builtin_amdgcn_cvt_pk_bf16_f32(lo, hi);
  return __builtin_bit_cast(unsigned, p);
#else
  unsigned r;
  asm("v_cvt_pk_bf16_f32 %0, %1, %2" : "=v"(r) : "v"(lo), "v"(hi));
  return r;
#endif
}

__device__ __forceinline__ float lrelu(float t) {
  return fmaxf(t, ALPHA_LRELU * t);   // exact leaky-relu for slope<1
}

// ---------------- phase 0: kernel[H,D,E] f32 -> kernelT[H,E,D] bf16 ----------------
__global__ void gat_prep_kernelT(const float* __restrict__ kernel,
                                 unsigned short* __restrict__ kT) {
  int idx = blockIdx.x * blockDim.x + threadIdx.x;
  if (idx >= GH * GE * GD) return;
  int d = idx % GD;
  int e = (idx / GD) % GE;
  int h = idx / (GD * GE);
  kT[idx] = f2bf(kernel[(h * GD + d) * GE + e]);
}

// ---------------- phase 1: feat = x @ W (WMMA bf16), a_s/a_n, featT bf16 ----------------
// One wave handles a 16-row x 128-col slab of feat for one (b,h).
__global__ void __launch_bounds__(128) gat_proj(
    const float* __restrict__ x,                // [B,N,D]
    const unsigned short* __restrict__ kT,      // [H,E,D] bf16
    const float* __restrict__ attn_self,        // [H,E]
    const float* __restrict__ attn_neigh,       // [H,E]
    unsigned short* __restrict__ featT,         // [B,H,E,N] bf16
    float* __restrict__ a_s,                    // [B,H,N]
    float* __restrict__ a_n)                    // [B,H,N]
{
  const int lane = threadIdx.x & 31;
  const int wid  = blockIdx.x * (blockDim.x >> 5) + (threadIdx.x >> 5);
  const int slab = wid & 63;            // N/16 slabs
  const int bh   = wid >> 6;
  const int h    = bh & (GH - 1);
  const int b    = bh >> 2;
  const int i0   = slab << 4;
  const int l15  = lane & 15;
  const int hi   = lane >> 4;           // which half of the wave
  const int off0 = hi << 3;

  v8f zero = {};
  v8f acc[8];
  #pragma unroll
  for (int s = 0; s < 8; ++s) acc[s] = zero;

  const float* xrow = x + ((size_t)b * GN + (i0 + l15)) * GD;

  #pragma unroll
  for (int kt = 0; kt < 4; ++kt) {
    const int k0 = kt << 5;
    // A operand: x rows, bf16, contiguous K layout (half0: K 0..15; half1: K 16..31)
    float4 f0 = *(const float4*)(xrow + k0 + (hi << 4));
    float4 f1 = *(const float4*)(xrow + k0 + (hi << 4) + 4);
    float4 f2 = *(const float4*)(xrow + k0 + (hi << 4) + 8);
    float4 f3 = *(const float4*)(xrow + k0 + (hi << 4) + 12);
    ABuf a;
    a.d[0] = pkbf(f0.x, f0.y); a.d[1] = pkbf(f0.z, f0.w);
    a.d[2] = pkbf(f1.x, f1.y); a.d[3] = pkbf(f1.z, f1.w);
    a.d[4] = pkbf(f2.x, f2.y); a.d[5] = pkbf(f2.z, f2.w);
    a.d[6] = pkbf(f3.x, f3.y); a.d[7] = pkbf(f3.z, f3.w);
    #pragma unroll
    for (int s = 0; s < 8; ++s) {
      // B operand: kernelT[h, e0+l15, k0 + hi*16 .. +15] -> 32 contiguous bytes
      BBuf bb;
      const uint4* bp = (const uint4*)(kT + (size_t)(h * GE + (s << 4) + l15) * GD + k0 + (hi << 4));
      bb.q[0] = bp[0];
      bb.q[1] = bp[1];
      acc[s] = __builtin_amdgcn_wmma_f32_16x16x32_bf16(
          false, a.v, false, bb.v, (short)0, acc[s], false, false);
    }
  }

  // store featT bf16: lane covers column e = s*16+l15; rows i0 + hi*8 + r
  const size_t fbase = (size_t)bh * GE * GN;
  #pragma unroll
  for (int s = 0; s < 8; ++s) {
    union { unsigned d[4]; uint4 q; } pk;
    #pragma unroll
    for (int j = 0; j < 4; ++j) pk.d[j] = pkbf(acc[s][2 * j], acc[s][2 * j + 1]);
    *(uint4*)(featT + fbase + (size_t)((s << 4) + l15) * GN + i0 + (hi << 3)) = pk.q;
  }

  // attention logits a_s/a_n: dot(feat_row, attn_vec) via shuffle reduction
  float wsv[8], wnv[8];
  #pragma unroll
  for (int s = 0; s < 8; ++s) {
    wsv[s] = attn_self [h * GE + (s << 4) + l15];
    wnv[s] = attn_neigh[h * GE + (s << 4) + l15];
  }
  #pragma unroll
  for (int r = 0; r < 8; ++r) {
    float ps = 0.f, pn = 0.f;
    #pragma unroll
    for (int s = 0; s < 8; ++s) { ps += acc[s][r] * wsv[s]; pn += acc[s][r] * wnv[s]; }
    #pragma unroll
    for (int m = 1; m < 16; m <<= 1) {   // stays within each 16-lane half
      ps += __shfl_xor(ps, m, 32);
      pn += __shfl_xor(pn, m, 32);
    }
    if (l15 == 0) {
      int row = i0 + (hi << 3) + r;
      a_s[(size_t)bh * GN + row] = ps;
      a_n[(size_t)bh * GN + row] = pn;
    }
  }
}

// ---------------- phase 2: fused masked-softmax + att@feat ----------------
// m = lrelu(a_s + max_j a_n) is a valid upper bound for every score in the row
// (leaky-relu is monotone), so no online rescaling and no adj pre-scan needed.
__global__ void __launch_bounds__(128) gat_attn(
    const float* __restrict__ adj,              // [B,N,N]
    const unsigned short* __restrict__ featT,   // [B,H,E,N] bf16
    const float* __restrict__ a_s,              // [B,H,N]
    const float* __restrict__ a_n,              // [B,H,N]
    const float* __restrict__ bias,             // [H,E]
    float* __restrict__ out)                    // [B,N,H*E]
{
  const int lane = threadIdx.x & 31;
  const int wid  = blockIdx.x * (blockDim.x >> 5) + (threadIdx.x >> 5);
  const int slab = wid & 63;
  const int bh   = wid >> 6;
  const int h    = bh & (GH - 1);
  const int b    = bh >> 2;
  const int i0   = slab << 4;
  const int l15  = lane & 15;
  const int hi   = lane >> 4;
  const int off0 = hi << 3;
  const int irow = i0 + l15;               // A-matrix row owned by this lane (both halves)

  const float asv = a_s[(size_t)bh * GN + irow];
  const float* adjrow = adj + ((size_t)b * GN + irow) * GN;
  const float* anrow  = a_n + (size_t)bh * GN;
  const unsigned short* fT = featT + (size_t)bh * GE * GN;

  // global max of a_n for this (b,h): 8 lane-strided float4 loads + wave reduction
  float mall = -__builtin_inff();
  {
    const float4* ap = (const float4*)anrow;
    #pragma unroll
    for (int t = 0; t < 8; ++t) {
      float4 v = ap[t * 32 + lane];
      mall = fmaxf(mall, fmaxf(fmaxf(v.x, v.y), fmaxf(v.z, v.w)));
    }
    #pragma unroll
    for (int m = 1; m < 32; m <<= 1) mall = fmaxf(mall, __shfl_xor(mall, m, 32));
  }
  const float mrow = lrelu(asv + mall);    // >= every score in this row

  v8f zero = {};
  v8f acc[8];
  #pragma unroll
  for (int s = 0; s < 8; ++s) acc[s] = zero;
  float lrow = 0.f;

  for (int jt = 0; jt < GN / 32; ++jt) {
    const int j0 = jt << 5;
    // a_n and adj for this lane's 16 K-slots (A-matrix layout order)
    float anv[16], adv[16];
    *(float4*)(anv + 0)  = *(const float4*)(anrow + j0 + (hi << 4));
    *(float4*)(anv + 4)  = *(const float4*)(anrow + j0 + (hi << 4) + 4);
    *(float4*)(anv + 8)  = *(const float4*)(anrow + j0 + (hi << 4) + 8);
    *(float4*)(anv + 12) = *(const float4*)(anrow + j0 + (hi << 4) + 12);
    *(float4*)(adv + 0)  = *(const float4*)(adjrow + j0 + (hi << 4));
    *(float4*)(adv + 4)  = *(const float4*)(adjrow + j0 + (hi << 4) + 4);
    *(float4*)(adv + 8)  = *(const float4*)(adjrow + j0 + (hi << 4) + 8);
    *(float4*)(adv + 12) = *(const float4*)(adjrow + j0 + (hi << 4) + 12);

    float p[16];
    float psum = 0.f;
    #pragma unroll
    for (int k = 0; k < 16; ++k) {
      float t = lrelu(asv + anv[k]) - mrow;
      float e = __expf(t) * adv[k];          // adj∈{0,1}: exact mask, exact passthrough
      p[k] = e;
      psum += e;
    }
    lrow += psum + __shfl_xor(psum, 16, 32); // halves see disjoint j's

    ABuf a;                                  // P tile in A-matrix bf16 layout
    #pragma unroll
    for (int j = 0; j < 8; ++j) a.d[j] = pkbf(p[2 * j], p[2 * j + 1]);

    #pragma unroll
    for (int s = 0; s < 8; ++s) {
      BBuf bb;   // featT[e, j0 + hi*16 .. +15]: 32 contiguous bytes, same contiguous K order as A
      const uint4* bp = (const uint4*)(fT + (size_t)((s << 4) + l15) * GN + j0 + (hi << 4));
      bb.q[0] = bp[0];
      bb.q[1] = bp[1];
      acc[s] = __builtin_amdgcn_wmma_f32_16x16x32_bf16(
          false, a.v, false, bb.v, (short)0, acc[s], false, false);
    }
  }

  // epilogue: normalize, bias, relu, head-major store out[b, row, h*E + e]
  #pragma unroll
  for (int r = 0; r < 8; ++r) {
    float lr  = __shfl(lrow, hi ? (r + 8) : r, 32);
    float inv = 1.0f / fmaxf(lr, 1e-30f);
    const int row = i0 + (hi << 3) + r;
    float* orow = out + ((size_t)b * GN + row) * (GH * GE) + h * GE;
    #pragma unroll
    for (int s = 0; s < 8; ++s) {
      const int e = (s << 4) + l15;
      float v = acc[s][r] * inv + bias[h * GE + e];
      orow[e] = fmaxf(v, 0.f);
    }
  }
}

extern "C" void kernel_launch(void* const* d_in, const int* in_sizes, int n_in,
                              void* d_out, int out_size, void* d_ws, size_t ws_size,
                              hipStream_t stream) {
  (void)in_sizes; (void)n_in; (void)out_size; (void)ws_size;
  const float* x          = (const float*)d_in[0];
  const float* adj        = (const float*)d_in[1];
  const float* kernel     = (const float*)d_in[2];
  const float* attn_self  = (const float*)d_in[3];
  const float* attn_neigh = (const float*)d_in[4];
  const float* bias       = (const float*)d_in[5];
  float* out = (float*)d_out;

  // workspace layout (~8.8 MB, fully rewritten every call)
  char* w = (char*)d_ws;
  const size_t featT_bytes = (size_t)GB * GH * GE * GN * 2;   // 8 MB bf16
  const size_t kT_bytes    = (size_t)GH * GE * GD * 2;        // 128 KB bf16
  const size_t an_bytes    = (size_t)GB * GH * GN * 4;        // 128 KB f32
  unsigned short* featT = (unsigned short*)w;
  unsigned short* kT    = (unsigned short*)(w + featT_bytes);
  float* a_s = (float*)(w + featT_bytes + kT_bytes);
  float* a_n = (float*)(w + featT_bytes + kT_bytes + an_bytes);

  gat_prep_kernelT<<<(GH * GE * GD + 255) / 256, 256, 0, stream>>>(kernel, kT);

  const int nwaves = GB * GH * (GN / 16);   // 2048 waves, 4 waves / block
  gat_proj<<<nwaves / 4, 128, 0, stream>>>(x, kT, attn_self, attn_neigh, featT, a_s, a_n);
  gat_attn<<<nwaves / 4, 128, 0, stream>>>(adj, featT, a_s, a_n, bias, out);
}
